// GraphConvEW_12627203850513
// MI455X (gfx1250) — matrix operand order repaired and checked by reading, not compiled, not run on hardware
//
#include <hip/hip_runtime.h>

#define NN 50000
#define NE 800000
#define NB 250
#define DIM 128

typedef __attribute__((ext_vector_type(16))) _Float16 v16h;
typedef __attribute__((ext_vector_type(8)))  _Float16 v8h;
typedef __attribute__((ext_vector_type(8)))  float    v8f;

union Frag16 { v16h v; v8h h[2]; };

// ---------------- small elementwise kernels ----------------

__global__ void zero_f32(float* __restrict__ p, int n) {
    int i = blockIdx.x * 256 + threadIdx.x;
    if (i < n) p[i] = 0.0f;
}

__global__ void edge_weight_kernel(const float* __restrict__ w, float* __restrict__ ws, int E) {
    int e = blockIdx.x * 256 + threadIdx.x;
    if (e < E) {
        ws[e] = 0.25f * (w[4 * e] + w[4 * e + 1] + w[4 * e + 2] + w[4 * e + 3]);
    }
}

__global__ void degree_kernel(const int* __restrict__ src, const int* __restrict__ dst,
                              float* __restrict__ dout, float* __restrict__ din, int E) {
    int e = blockIdx.x * 256 + threadIdx.x;
    if (e < E) {
        atomicAdd(dout + src[e], 1.0f);
        atomicAdd(din  + dst[e], 1.0f);
    }
}

// in-place: degree -> rsqrt(max(deg,1))
__global__ void norm_kernel(float* __restrict__ deg, int n) {
    int i = blockIdx.x * 256 + threadIdx.x;
    if (i < n) deg[i] = rsqrtf(fmaxf(deg[i], 1.0f));
}

// W [K=128, N=128] fp32 row-major -> Wt [N,K] f16 (transposed, K contiguous)
__global__ void wt_convert(const float* __restrict__ W, _Float16* __restrict__ Wt) {
    int i = blockIdx.x * 256 + threadIdx.x;   // i = k*128 + n
    int k = i >> 7, n = i & 127;
    Wt[n * DIM + k] = (_Float16)W[i];
}

// Hb = f16(agg * in_norm[row])
__global__ void scale_to_f16(const float* __restrict__ agg, const float* __restrict__ inorm,
                             _Float16* __restrict__ hb) {
    int i = blockIdx.x * 256 + threadIdx.x;
    hb[i] = (_Float16)(agg[i] * inorm[i >> 7]);
}

// ---------------- edge gather / scatter-add ----------------
// one wave per edge; lane loads float4 of h[src], scatter-adds into agg[dst] (L2 atomics)
__global__ void edge_scatter(const float* __restrict__ h, const float* __restrict__ ws,
                             const float* __restrict__ onorm,
                             const int* __restrict__ src, const int* __restrict__ dst,
                             float* __restrict__ agg) {
    int e    = blockIdx.x * 8 + (threadIdx.x >> 5);
    int lane = threadIdx.x & 31;
    int s = src[e];
    int d = dst[e];
    float coef = ws[e] * onorm[s];
    float4 v = ((const float4*)(h + (size_t)s * DIM))[lane];
    float* ap = agg + (size_t)d * DIM + lane * 4;
    atomicAdd(ap + 0, v.x * coef);
    atomicAdd(ap + 1, v.y * coef);
    atomicAdd(ap + 2, v.z * coef);
    atomicAdd(ap + 3, v.w * coef);
}

// ---------------- WMMA GEMM + fused relu + LayerNorm ----------------
// block = 256 threads = 8 waves; block covers 16 rows x 128 cols; wave w -> cols [16w,16w+16)
__global__ void gemm_relu_ln(const _Float16* __restrict__ Hb, const _Float16* __restrict__ Wt,
                             const float* __restrict__ g, const float* __restrict__ b,
                             float* __restrict__ out) {
    __shared__ float tile[16][132];      // padded: rows r / r+8 are 32 banks apart
    __shared__ float mu_s[16], rs_s[16];

    const int lane = threadIdx.x & 31;
    const int wave = threadIdx.x >> 5;
    const int m0   = blockIdx.x * 16;
    const int n0   = wave * 16;
    const int low  = lane & 15;          // A: row index; B: col index
    const int hi   = lane >> 4;          // 0 or 1
    const int akb  = hi * 8;             // A k-base (ISA 16-bit A 16x32 layout)
    const int bkb  = hi * 16;            // unused: B loads use akb, same k order as A

    const _Float16* arow = Hb + (size_t)(m0 + low) * DIM;
    const _Float16* brow = Wt + (size_t)(n0 + low) * DIM;

    v8f acc = {};
#pragma unroll
    for (int k0 = 0; k0 < DIM; k0 += 32) {
        Frag16 a, bf;
        a.h[0]  = *(const v8h*)(arow + k0 + akb);        // K = k0+akb .. +7
        a.h[1]  = *(const v8h*)(arow + k0 + 16 + akb);   // K = k0+16+akb .. +7
        bf.h[0] = *(const v8h*)(brow + k0 + akb);        // K = k0+akb .. +7
        bf.h[1] = *(const v8h*)(brow + k0 + 16 + akb);   // K = k0+16+akb .. +7
        acc = __builtin_amdgcn_wmma_f32_16x16x32_f16(
            false, a.v, false, bf.v, (short)0, acc, false, false);
    }

    // relu, stage to LDS (C/D layout: VGPR r -> row r + 8*hi, col = n0 + low)
#pragma unroll
    for (int r = 0; r < 8; ++r) {
        tile[r + hi * 8][n0 + low] = fmaxf(acc[r], 0.0f);
    }
    __syncthreads();

    if (threadIdx.x < 16) {
        float s = 0.0f, ss = 0.0f;
        for (int c = 0; c < DIM; ++c) {
            float v = tile[threadIdx.x][c];
            s += v; ss += v * v;
        }
        float mu  = s * (1.0f / DIM);
        float var = ss * (1.0f / DIM) - mu * mu;
        mu_s[threadIdx.x] = mu;
        rs_s[threadIdx.x] = rsqrtf(var + 1e-5f);
    }
    __syncthreads();

    for (int idx = threadIdx.x; idx < 16 * DIM; idx += 256) {
        int row = idx >> 7, col = idx & 127;
        float v = tile[row][col];
        out[(size_t)(m0 + row) * DIM + col] =
            (v - mu_s[row]) * rs_s[row] * g[col] + b[col];
    }
}

// ---------------- WMMA GEMM + fused per-graph readout ----------------
__global__ void gemm_readout(const _Float16* __restrict__ Hb, const _Float16* __restrict__ Wt,
                             const int* __restrict__ gids, float* __restrict__ out) {
    const int lane = threadIdx.x & 31;
    const int wave = threadIdx.x >> 5;
    const int m0   = blockIdx.x * 16;
    const int n0   = wave * 16;
    const int low  = lane & 15;
    const int hi   = lane >> 4;
    const int akb  = hi * 8;
    const int bkb  = hi * 16;

    const _Float16* arow = Hb + (size_t)(m0 + low) * DIM;
    const _Float16* brow = Wt + (size_t)(n0 + low) * DIM;

    v8f acc = {};
#pragma unroll
    for (int k0 = 0; k0 < DIM; k0 += 32) {
        Frag16 a, bf;
        a.h[0]  = *(const v8h*)(arow + k0 + akb);
        a.h[1]  = *(const v8h*)(arow + k0 + 16 + akb);
        bf.h[0] = *(const v8h*)(brow + k0 + akb);
        bf.h[1] = *(const v8h*)(brow + k0 + 16 + akb);
        acc = __builtin_amdgcn_wmma_f32_16x16x32_f16(
            false, a.v, false, bf.v, (short)0, acc, false, false);
    }

#pragma unroll
    for (int r = 0; r < 8; ++r) {
        int row = m0 + r + hi * 8;
        int gid = gids[row];
        atomicAdd(out + (size_t)gid * DIM + n0 + low, acc[r]);
    }
}

// ---------------- launch ----------------

extern "C" void kernel_launch(void* const* d_in, const int* in_sizes, int n_in,
                              void* d_out, int out_size, void* d_ws, size_t ws_size,
                              hipStream_t stream) {
    const float* x     = (const float*)d_in[0];
    const float* w     = (const float*)d_in[1];
    const float* W1    = (const float*)d_in[2];
    const float* W2    = (const float*)d_in[3];
    const float* W3    = (const float*)d_in[4];
    const float* ln1_g = (const float*)d_in[5];
    const float* ln1_b = (const float*)d_in[6];
    const float* ln2_g = (const float*)d_in[7];
    const float* ln2_b = (const float*)d_in[8];
    const int*   src   = (const int*)d_in[9];
    const int*   dst   = (const int*)d_in[10];
    const int*   gids  = (const int*)d_in[11];
    float* out = (float*)d_out;

    // workspace layout (bytes)
    char* ws = (char*)d_ws;
    float*    ws_e   = (float*)(ws + 0);                       // E floats
    float*    onorm  = (float*)(ws + 3200000);                 // N floats (deg_out -> rsqrt)
    float*    inorm  = (float*)(ws + 3400000);                 // N floats (deg_in -> rsqrt)
    float*    agg    = (float*)(ws + 3600000);                 // N*128 f32
    float*    h      = (float*)(ws + 29200000);                // N*128 f32
    _Float16* hb     = (_Float16*)(ws + 54800000);             // N*128 f16
    _Float16* Wt1    = (_Float16*)(ws + 67600000);             // 128*128 f16
    _Float16* Wt2    = (_Float16*)(ws + 67632768);
    _Float16* Wt3    = (_Float16*)(ws + 67665536);

    const int NB_NODE   = (NN + 255) / 256;        // 196
    const int NB_EDGE   = NE / 256;                // 3125
    const int NB_FEAT   = (NN * DIM) / 256;        // 25000
    const int NB_GEMM   = NN / 16;                 // 3125
    const int NB_SCAT   = NE / 8;                  // 100000
    const int NB_OUT    = (NB * DIM + 255) / 256;  // 125
    const int NB_W      = (DIM * DIM) / 256;       // 64

    // --- preprocessing ---
    zero_f32<<<NB_NODE, 256, 0, stream>>>(onorm, NN);
    zero_f32<<<NB_NODE, 256, 0, stream>>>(inorm, NN);
    zero_f32<<<NB_OUT,  256, 0, stream>>>(out, NB * DIM);
    edge_weight_kernel<<<NB_EDGE, 256, 0, stream>>>(w, ws_e, NE);
    degree_kernel<<<NB_EDGE, 256, 0, stream>>>(src, dst, onorm, inorm, NE);
    norm_kernel<<<NB_NODE, 256, 0, stream>>>(onorm, NN);
    norm_kernel<<<NB_NODE, 256, 0, stream>>>(inorm, NN);
    wt_convert<<<NB_W, 256, 0, stream>>>(W1, Wt1);
    wt_convert<<<NB_W, 256, 0, stream>>>(W2, Wt2);
    wt_convert<<<NB_W, 256, 0, stream>>>(W3, Wt3);

    // --- layer 1 ---
    zero_f32<<<NB_FEAT, 256, 0, stream>>>(agg, NN * DIM);
    edge_scatter<<<NB_SCAT, 256, 0, stream>>>(x, ws_e, onorm, src, dst, agg);
    scale_to_f16<<<NB_FEAT, 256, 0, stream>>>(agg, inorm, hb);
    gemm_relu_ln<<<NB_GEMM, 256, 0, stream>>>(hb, Wt1, ln1_g, ln1_b, h);

    // --- layer 2 ---
    zero_f32<<<NB_FEAT, 256, 0, stream>>>(agg, NN * DIM);
    edge_scatter<<<NB_SCAT, 256, 0, stream>>>(h, ws_e, onorm, src, dst, agg);
    scale_to_f16<<<NB_FEAT, 256, 0, stream>>>(agg, inorm, hb);
    gemm_relu_ln<<<NB_GEMM, 256, 0, stream>>>(hb, Wt2, ln2_g, ln2_b, h);

    // --- layer 3 + readout ---
    zero_f32<<<NB_FEAT, 256, 0, stream>>>(agg, NN * DIM);
    edge_scatter<<<NB_SCAT, 256, 0, stream>>>(h, ws_e, onorm, src, dst, agg);
    scale_to_f16<<<NB_FEAT, 256, 0, stream>>>(agg, inorm, hb);
    gemm_readout<<<NB_GEMM, 256, 0, stream>>>(hb, Wt3, gids, out);

    (void)in_sizes; (void)n_in; (void)out_size; (void)ws_size;
}
